// DownSampleHead_70514773066128
// MI455X (gfx1250) — matrix operand order repaired and checked by reading, not compiled, not run on hardware
//
#include <hip/hip_runtime.h>
#include <hip/hip_bf16.h>

// ---------------------------------------------------------------------------
// DownSampleHead for MI455X (gfx1250): BN folded into bf16 WMMA GEMMs.
//   bn_stats -> fold(BN into W, bf16) -> WMMA GEMM (+act)
//   A-tile LDS fill uses gfx1250 global_load_async_to_lds_b128 (ASYNCcnt).
//   cls conv2 via WMMA (200 rows zero-padded to 256); bg/reg conv2 tiny VALU.
// ---------------------------------------------------------------------------

typedef __attribute__((ext_vector_type(16))) __bf16 v16bf;
typedef __attribute__((ext_vector_type(8)))  __bf16 v8bf;
typedef __attribute__((ext_vector_type(8)))  float  v8f;

#define EPS 1e-5f
#define ACT_RELU 0
#define ACT_TANH 1
#define ACT_NONE 2

#define BM 128
#define BN 64
#define BK 32    // K per WMMA slab
#define BKO 64   // K per main-loop iteration (2 slabs per barrier pair)
#define LDA 40   // bf16 elements per LDS row (80B: 16B-aligned, bank-spread)
#define LDB 40

static __device__ __forceinline__ __bf16 f2bf(float f) { return (__bf16)f; }

static __device__ __forceinline__ unsigned packbf(float a, float b) {
  unsigned short ua = __builtin_bit_cast(unsigned short, (__bf16)a);
  unsigned short ub = __builtin_bit_cast(unsigned short, (__bf16)b);
  return (unsigned)ua | ((unsigned)ub << 16);
}

static __device__ __forceinline__ v16bf combine8(v8bf lo, v8bf hi) {
  return __builtin_shufflevector(lo, hi, 0,1,2,3,4,5,6,7,8,9,10,11,12,13,14,15);
}

// async 32B contiguous copy global->LDS (two b128 ops, ASYNCcnt-tracked)
static __device__ __forceinline__ void async_copy32(const __bf16* g, __bf16* l) {
  unsigned lds = (unsigned)(size_t)l;                    // LDS aperture: low 32b = offset
  unsigned long long ga = (unsigned long long)(size_t)g;
  asm volatile("global_load_async_to_lds_b128 %0, %1, off\n\t"
               "global_load_async_to_lds_b128 %0, %1, off offset:16"
               :: "v"(lds), "v"(ga) : "memory");
}
static __device__ __forceinline__ void async_wait0() {
  asm volatile("s_wait_asynccnt 0" ::: "memory");
}

// ---------------------------------------------------------------------------
// Per-channel mean/var over (B, T) of x[B][C][T]. One block per channel.
// ---------------------------------------------------------------------------
__global__ __launch_bounds__(256) void bn_stats_kernel(
    const float* __restrict__ x, float* __restrict__ mean, float* __restrict__ var,
    int C, int T, int Bn) {
  __shared__ float rs[256], rq[256];
  int c = blockIdx.x, tid = threadIdx.x;
  int n = Bn * T;
  float s = 0.f, q = 0.f;
  for (int i = tid; i < n; i += 256) {
    int b = i / T, t = i - b * T;
    float v = x[((size_t)b * C + c) * T + t];
    s += v; q += v * v;
  }
  rs[tid] = s; rq[tid] = q;
  __syncthreads();
  for (int off = 128; off > 0; off >>= 1) {
    if (tid < off) { rs[tid] += rs[tid + off]; rq[tid] += rq[tid + off]; }
    __syncthreads();
  }
  if (tid == 0) {
    float m = rs[0] / (float)n;
    mean[c] = m;
    var[c] = rq[0] / (float)n - m * m;  // biased var, matches jnp.var
  }
}

// ---------------------------------------------------------------------------
// Fold BN into conv weight: Weff[o,c] = bf16(W[o,c]*s[c]),
// beff[o] = b[o] + sum_c W[o,c]*h[c].  One block per output row o.
// ---------------------------------------------------------------------------
__global__ __launch_bounds__(256) void fold_kernel(
    const float* __restrict__ W, const float* __restrict__ bias,
    const float* __restrict__ gamma, const float* __restrict__ beta,
    const float* __restrict__ mean, const float* __restrict__ var,
    __bf16* __restrict__ Weff, float* __restrict__ beff, int C, int Mvalid) {
  __shared__ float red[256];
  int o = blockIdx.x, tid = threadIdx.x;
  if (o >= Mvalid) {
    for (int c = tid; c < C; c += 256) Weff[(size_t)o * C + c] = f2bf(0.f);
    if (tid == 0) beff[o] = 0.f;
    return;
  }
  float acc = 0.f;
  for (int c = tid; c < C; c += 256) {
    float s = gamma[c] * rsqrtf(var[c] + EPS);
    float h = beta[c] - mean[c] * s;
    float w = W[(size_t)o * C + c];
    Weff[(size_t)o * C + c] = f2bf(w * s);
    acc += w * h;
  }
  red[tid] = acc;
  __syncthreads();
  for (int off = 128; off > 0; off >>= 1) {
    if (tid < off) red[tid] += red[tid + off];
    __syncthreads();
  }
  if (tid == 0) beff[o] = bias[o] + red[0];
}

// f32 -> bf16 weight convert with zero-padding (for cls conv2, 200 -> 256 rows)
__global__ __launch_bounds__(256) void cvt_pad_kernel(
    const float* __restrict__ W, __bf16* __restrict__ Wb, int C, int Mvalid, int Mpad) {
  int i = blockIdx.x * 256 + threadIdx.x;
  if (i >= Mpad * C) return;
  int o = i / C;
  Wb[i] = f2bf(o < Mvalid ? W[i] : 0.f);
}

// ---------------------------------------------------------------------------
// WMMA GEMM: Y[b, o, t0+n] = act( sum_k Aeff[o,k] * X[b,k,t0+n] + bias[o] )
// A: bf16 [Mpad][K] row-major (async-copied to LDS).
// X: f32 [B][K][T] (converted to bf16 at LDS fill, stored transposed [n][k]).
// Block tile 128x64, K-unroll 64 (2 slabs of 32), 8 waves of 32x32,
// v_wmma_f32_16x16x32_bf16.  grid = (Mpad/BM, (32*T)/BN); T % 64 == 0.
// ---------------------------------------------------------------------------
__global__ __launch_bounds__(256) void gemm_wmma_kernel(
    const __bf16* __restrict__ A, const float* __restrict__ bias,
    const float* __restrict__ X, float* __restrict__ Y,
    int K, int T, int Mvalid, int OCout, int act) {
  __shared__ alignas(16) __bf16 As[2][BM * LDA];
  __shared__ alignas(16) __bf16 Bs[2][BN * LDB];

  const int tid = threadIdx.x;
  const int m0 = blockIdx.x * BM;
  const int n0 = blockIdx.y * BN;
  const int b = n0 / T;
  const int t0 = n0 - b * T;
  const float* __restrict__ Xb = X + (size_t)b * K * T;

  const int wave = tid >> 5, lane = tid & 31;
  const int wm = (wave & 3) * 32;   // 4 waves along M
  const int wn = (wave >> 2) * 32;  // 2 waves along N
  const int lrow = lane & 15;
  const int lsel = lane >> 4;

  v8f acc[2][2] = {};

  // fill mappings
  const int ra = tid >> 1;            // A row 0..127
  const int kA = (tid & 1) * 16;      // 16 bf16 (32B) per thread per slab
  const int ng = (tid & 15) * 4;      // B: 4 contiguous t per thread
  const int kp = (tid >> 4) * 2;      // B: K-pair base within slab

  for (int k0 = 0; k0 < K; k0 += BKO) {
    __syncthreads();
#pragma unroll
    for (int s = 0; s < 2; ++s) {
      // --- A tile slab s: async contiguous copy global->LDS ---
      const __bf16* ag = A + (size_t)(m0 + ra) * K + k0 + s * BK + kA;
      async_copy32(ag, &As[s][ra * LDA + kA]);
      // --- X tile slab s: 2 adjacent-K rows x 4 t, cvt+pack, transposed store
      const float* xg0 = Xb + (size_t)(k0 + s * BK + kp) * T + t0 + ng;
      float4 r0 = *(const float4*)(xg0);
      float4 r1 = *(const float4*)(xg0 + T);
      float a0[4] = {r0.x, r0.y, r0.z, r0.w};
      float a1[4] = {r1.x, r1.y, r1.z, r1.w};
#pragma unroll
      for (int j = 0; j < 4; ++j)
        *(unsigned*)&Bs[s][(ng + j) * LDB + kp] = packbf(a0[j], a1[j]);
      if (k0 + BKO < K)  // prefetch next iteration's X rows
        __builtin_prefetch(xg0 + (size_t)BKO * T, 0, 1);
    }
    async_wait0();
    __syncthreads();

#pragma unroll
    for (int s = 0; s < 2; ++s) {
      v16bf afr[2], bfr[2];
#pragma unroll
      for (int i = 0; i < 2; ++i) {
        int am = wm + i * 16 + lrow;
        v8bf lo = *(const v8bf*)(&As[s][am * LDA + lsel * 8]);       // K 0..7 / 8..15
        v8bf hi = *(const v8bf*)(&As[s][am * LDA + lsel * 8 + 16]);  // K 16..23 / 24..31
        afr[i] = combine8(lo, hi);
      }
#pragma unroll
      for (int j = 0; j < 2; ++j) {
        int bn = wn + j * 16 + lrow;
        v8bf lo = *(const v8bf*)(&Bs[s][bn * LDB + lsel * 8]);
        v8bf hi = *(const v8bf*)(&Bs[s][bn * LDB + lsel * 8 + 16]);
        bfr[j] = combine8(lo, hi);
      }
#pragma unroll
      for (int i = 0; i < 2; ++i)
#pragma unroll
        for (int j = 0; j < 2; ++j)
          acc[i][j] = __builtin_amdgcn_wmma_f32_16x16x32_bf16(
              false, afr[i], false, bfr[j], (short)0, acc[i][j], false, false);
    }
  }

  // --- epilogue: bias + activation, masked store ---
#pragma unroll
  for (int i = 0; i < 2; ++i) {
#pragma unroll
    for (int j = 0; j < 2; ++j) {
      int obase = m0 + wm + i * 16 + lsel * 8;
      int tcol = t0 + wn + j * 16 + lrow;
#pragma unroll
      for (int v = 0; v < 8; ++v) {
        int o = obase + v;
        if (o < Mvalid) {
          float r = acc[i][j][v] + bias[o];
          if (act == ACT_RELU) r = fmaxf(r, 0.f);
          else if (act == ACT_TANH) r = tanhf(r);
          Y[((size_t)b * OCout + o) * T + tcol] = r;
        }
      }
    }
  }
}

// ---------------------------------------------------------------------------
// bg conv2: out[b,200,t] = sigmoid( sum_c W2[c]*H[b,c,t] + b2 ), IC=256
// ---------------------------------------------------------------------------
__global__ __launch_bounds__(256) void bg_conv2_kernel(
    const float* __restrict__ H, const float* __restrict__ W2,
    const float* __restrict__ b2, float* __restrict__ Ycls, int T) {
  int i = blockIdx.x * 256 + threadIdx.x;
  if (i >= 32 * T) return;
  int b = i / T, t = i - b * T;
  const float* h = H + (size_t)b * 256 * T + t;
  float s = b2[0];
  for (int c = 0; c < 256; ++c) s += W2[c] * h[(size_t)c * T];
  Ycls[((size_t)b * 201 + 200) * T + t] = 1.f / (1.f + __expf(-s));
}

// ---------------------------------------------------------------------------
// reg conv2: out[b,o,t] = scale * tanh( sum_c W2[o,c]*H[b,c,t] + b2[o] ), IC=512
// ---------------------------------------------------------------------------
__global__ __launch_bounds__(256) void reg_conv2_kernel(
    const float* __restrict__ H, const float* __restrict__ W2,
    const float* __restrict__ b2, const float* __restrict__ scales, int level,
    float* __restrict__ Yreg, int T) {
  int i = blockIdx.x * 256 + threadIdx.x;
  if (i >= 32 * T) return;
  int b = i / T, t = i - b * T;
  const float* h = H + (size_t)b * 512 * T + t;
  float sc = scales[level];
  float s0 = b2[0], s1 = b2[1], s2 = b2[2];
  for (int c = 0; c < 512; ++c) {
    float v = h[(size_t)c * T];
    s0 += W2[c] * v;
    s1 += W2[512 + c] * v;
    s2 += W2[1024 + c] * v;
  }
  Yreg[((size_t)b * 3 + 0) * T + t] = sc * tanhf(s0);
  Yreg[((size_t)b * 3 + 1) * T + t] = sc * tanhf(s1);
  Yreg[((size_t)b * 3 + 2) * T + t] = sc * tanhf(s2);
}

// ---------------------------------------------------------------------------
extern "C" void kernel_launch(void* const* d_in, const int* in_sizes, int n_in,
                              void* d_out, int out_size, void* d_ws, size_t ws_size,
                              hipStream_t stream) {
  static const int Ts[5] = {1024, 512, 256, 128, 64};
  // --- input pointer map (jax pytree order: dict keys sorted) ---
  const float* feats[5];
  for (int l = 0; l < 5; ++l) feats[l] = (const float*)d_in[l];
  // bg: 5..10  (W1, W2, b1, b2, beta, gamma)
  const float* bgW1 = (const float*)d_in[5];
  const float* bgW2 = (const float*)d_in[6];
  const float* bgB1 = (const float*)d_in[7];
  const float* bgB2 = (const float*)d_in[8];
  const float* bgBe = (const float*)d_in[9];
  const float* bgGa = (const float*)d_in[10];
  // cls: 11..16
  const float* clW1 = (const float*)d_in[11];
  const float* clW2 = (const float*)d_in[12];
  const float* clB1 = (const float*)d_in[13];
  const float* clB2 = (const float*)d_in[14];
  const float* clBe = (const float*)d_in[15];
  const float* clGa = (const float*)d_in[16];
  // reg: 17..22
  const float* rgW1 = (const float*)d_in[17];
  const float* rgW2 = (const float*)d_in[18];
  const float* rgB1 = (const float*)d_in[19];
  const float* rgB2 = (const float*)d_in[20];
  const float* rgBe = (const float*)d_in[21];
  const float* rgGa = (const float*)d_in[22];
  const float* scales = (const float*)d_in[23];
  // trunc[l]: 24+4l .. (W, b, beta, gamma)
  const float *tW[5], *tB[5], *tBe[5], *tGa[5];
  for (int l = 0; l < 5; ++l) {
    tW[l]  = (const float*)d_in[24 + 4 * l];
    tB[l]  = (const float*)d_in[25 + 4 * l];
    tBe[l] = (const float*)d_in[26 + 4 * l];
    tGa[l] = (const float*)d_in[27 + 4 * l];
  }

  // --- workspace carve (bytes) ---
  char* ws = (char*)d_ws;
  size_t off = 0;
  auto carve = [&](size_t bytes) {
    char* p = ws + off;
    off = (off + bytes + 255) & ~(size_t)255;
    return p;
  };
  float*  outs  = (float*)carve(32ull * 512 * 1024 * 4);   // trunk outputs (max level 0)
  float*  Hbuf  = (float*)carve(32ull * 512 * 1024 * 4);   // head hidden
  __bf16* WeffT = (__bf16*)carve(512ull * 1024 * 2);       // trunk folded W
  float*  beffT = (float*)carve(512 * 4);
  __bf16* W1eff = (__bf16*)carve(512ull * 512 * 2);        // head conv1 folded W
  float*  b1eff = (float*)carve(512 * 4);
  __bf16* W2cb  = (__bf16*)carve(256ull * 512 * 2);        // cls conv2 bf16, padded
  float*  meanB = (float*)carve(1024 * 4);
  float*  varB  = (float*)carve(1024 * 4);
  (void)ws_size; (void)in_sizes; (void)n_in; (void)out_size;

  // output block offsets: 5x [B,201,T] then 5x [B,3,T]
  size_t clsOff[5], regOff[5], acc = 0;
  for (int l = 0; l < 5; ++l) { clsOff[l] = acc; acc += 32ull * 201 * Ts[l]; }
  for (int l = 0; l < 5; ++l) { regOff[l] = acc; acc += 32ull * 3 * Ts[l]; }
  float* out = (float*)d_out;

  // one-time: cls conv2 weights -> bf16, zero-padded 200 -> 256 rows
  cvt_pad_kernel<<<(256 * 512) / 256, 256, 0, stream>>>(clW2, W2cb, 512, 200, 256);

  for (int l = 0; l < 5; ++l) {
    const int T = Ts[l];
    const dim3 blk(256);
    const int nTilesT = (32 * T) / BN;

    // ---- trunk: BN stats -> fold -> GEMM(ReLU) 1024->512 ----
    bn_stats_kernel<<<1024, blk, 0, stream>>>(feats[l], meanB, varB, 1024, T, 32);
    fold_kernel<<<512, blk, 0, stream>>>(tW[l], tB[l], tGa[l], tBe[l], meanB, varB,
                                         WeffT, beffT, 1024, 512);
    gemm_wmma_kernel<<<dim3(512 / BM, nTilesT), blk, 0, stream>>>(
        WeffT, beffT, feats[l], outs, 1024, T, 512, 512, ACT_RELU);

    // ---- shared head BN stats over outs ----
    bn_stats_kernel<<<512, blk, 0, stream>>>(outs, meanB, varB, 512, T, 32);

    // ---- cls head: conv1(tanh) 512->512, conv2 512->200 (+bias) ----
    fold_kernel<<<512, blk, 0, stream>>>(clW1, clB1, clGa, clBe, meanB, varB,
                                         W1eff, b1eff, 512, 512);
    gemm_wmma_kernel<<<dim3(512 / BM, nTilesT), blk, 0, stream>>>(
        W1eff, b1eff, outs, Hbuf, 512, T, 512, 512, ACT_TANH);
    gemm_wmma_kernel<<<dim3(256 / BM, nTilesT), blk, 0, stream>>>(
        W2cb, clB2, Hbuf, out + clsOff[l], 512, T, 200, 201, ACT_NONE);

    // ---- bg head: conv1(tanh) 512->256, conv2 256->1 (sigmoid) ----
    fold_kernel<<<256, blk, 0, stream>>>(bgW1, bgB1, bgGa, bgBe, meanB, varB,
                                         W1eff, b1eff, 512, 256);
    gemm_wmma_kernel<<<dim3(256 / BM, nTilesT), blk, 0, stream>>>(
        W1eff, b1eff, outs, Hbuf, 512, T, 256, 256, ACT_TANH);
    bg_conv2_kernel<<<(32 * T + 255) / 256, blk, 0, stream>>>(
        Hbuf, bgW2, bgB2, out + clsOff[l], T);

    // ---- reg head: conv1(tanh) 512->512, conv2 512->3 (scale*tanh) ----
    fold_kernel<<<512, blk, 0, stream>>>(rgW1, rgB1, rgGa, rgBe, meanB, varB,
                                         W1eff, b1eff, 512, 512);
    gemm_wmma_kernel<<<dim3(512 / BM, nTilesT), blk, 0, stream>>>(
        W1eff, b1eff, outs, Hbuf, 512, T, 512, 512, ACT_TANH);
    reg_conv2_kernel<<<(32 * T + 255) / 256, blk, 0, stream>>>(
        Hbuf, rgW2, rgB2, scales, l, out + regOff[l], T);
  }
}
